// BlockMLP_27504970564323
// MI455X (gfx1250) — compile-verified, matrix operand order checked by reading, not run on hardware
//
#include <hip/hip_runtime.h>
#include <hip/hip_bf16.h>
#include <stdint.h>

// ---------------------------------------------------------------------------
// BlockMLP fused kernel for gfx1250 (MI455X), bf16 WMMA path, b128 LDS ops,
// double-buffered LDS->WMMA pipeline, immediate-offset output stores.
//   x:[8192,4096] f32, W1:[64,64,256], b1:[64,1,256], W2:[64,256,64], b2:[64,1,64]
//   out:[8192,4096] f32
// ---------------------------------------------------------------------------

typedef __attribute__((ext_vector_type(16))) __bf16 v16bf;
typedef __attribute__((ext_vector_type(8)))  float  v8f;

#define N_BLOCKS   64
#define BLOCK_DIM  64
#define HIDDEN     256
#define IN_DIM     4096
#define BATCH      8192
#define M_TILE     64          // rows per inner tile
#define M_SUPER    512         // rows per workgroup
#define N_SUB      (M_SUPER / M_TILE)

// LDS row strides in bf16 elements. Multiples of 8 elements (16 B) so every
// fragment chunk is 16B-aligned -> ds_load_b128.  Bank patterns:
//   72 el = 144 B = 36 banks/row  -> 16 b128 starts all distinct mod 64 banks
//  264 el = 528 B =  4 banks/row  -> 16 b128 starts all distinct mod 64 banks
#define XS_STRIDE  72          // for K=64 arrays (x tile, W1^T)
#define HS_STRIDE  264         // for K=256 arrays (h tile, W2^T)

#define OFF_W1T 0
#define SZ_W1T  (HIDDEN * XS_STRIDE * 2)        // [256][72] bf16 (W1^T: [out][in])
#define OFF_W2T (OFF_W1T + SZ_W1T)
#define SZ_W2T  (BLOCK_DIM * HS_STRIDE * 2)     // [64][264] bf16 (W2^T: [out][in])
#define OFF_XS  (OFF_W2T + SZ_W2T)
#define SZ_XS   (M_TILE * XS_STRIDE * 2)        // [64][72] bf16
#define OFF_HS  (OFF_XS + SZ_XS)
#define SZ_HS   (M_TILE * HS_STRIDE * 2)        // [64][264] bf16
#define OFF_B1  (OFF_HS + SZ_HS)
#define SZ_B1   (HIDDEN * 4)
#define OFF_B2  (OFF_B1 + SZ_B1)
#define SZ_B2   (BLOCK_DIM * 4)
#define SMEM_BYTES (OFF_B2 + SZ_B2)             // ~112 KB < 320 KB WGP LDS

union Frag {
    v16bf v;
    uint4 q[2];
};

// Pack two f32 -> packed bf16 dword.
static __device__ __forceinline__ uint32_t pkbf(float a, float b) {
    union { __bf16 h[2]; uint32_t u; } z;
    z.h[0] = (__bf16)a;
    z.h[1] = (__bf16)b;
    return z.u;
}

static __device__ __forceinline__ float gelu_exact(float t) {
    return 0.5f * t * (1.0f + erff(t * 0.70710678118654752440f));
}

// Load a 16x32 bf16 WMMA operand fragment (A: row-major MxK, or B via
// transposed [N][K] storage) from LDS, per ISA 7.12.2 striping:
//   lane L, element e : major = major0 + (L%16),  K = kbase + (e/8)*16 + (L/16)*8 + (e%8)
// Both 8-element chunks are 16B-aligned -> two ds_load_b128.
static __device__ __forceinline__ void load_frag(Frag& f, const __bf16* base,
                                                 int major0, int stride, int kbase,
                                                 int lr, int lh) {
    const __bf16* p = base + (major0 + lr) * stride + kbase + lh * 8;
    f.q[0] = *(const uint4*)p;          // K = kbase +    lh*8 .. +7
    f.q[1] = *(const uint4*)(p + 16);   // K = kbase+16 + lh*8 .. +7
}

__global__ __launch_bounds__(256, 2)
void blockmlp_wmma_kernel(const float* __restrict__ x,
                          const float* __restrict__ W1,
                          const float* __restrict__ b1,
                          const float* __restrict__ W2,
                          const float* __restrict__ b2,
                          float* __restrict__ out) {
    extern __shared__ char smem[];
    __bf16* w1t = (__bf16*)(smem + OFF_W1T);
    __bf16* w2t = (__bf16*)(smem + OFF_W2T);
    __bf16* xs  = (__bf16*)(smem + OFF_XS);
    __bf16* hs  = (__bf16*)(smem + OFF_HS);
    float*  b1s = (float*)(smem + OFF_B1);
    float*  b2s = (float*)(smem + OFF_B2);

    const int tid  = threadIdx.x;
    const int wave = tid >> 5;
    const int lane = tid & 31;
    const int lr   = lane & 15;   // row/col within 16
    const int lh   = lane >> 4;   // K half select

    const int blk   = blockIdx.y;              // 0..63 block index
    const int mbase = blockIdx.x * M_SUPER;    // first row of this workgroup

    // ---- Stage weights + biases into LDS (transposed, bf16), once ----
    const float* W1b = W1 + (size_t)blk * BLOCK_DIM * HIDDEN;   // [64][256]
    const float* W2b = W2 + (size_t)blk * HIDDEN * BLOCK_DIM;   // [256][64]
    for (int t = tid; t < BLOCK_DIM * HIDDEN; t += 256) {
        int i = t >> 8, o = t & 255;                  // W1[i][o]
        w1t[o * XS_STRIDE + i] = (__bf16)W1b[t];
    }
    for (int t = tid; t < HIDDEN * BLOCK_DIM; t += 256) {
        int k = t >> 6, o = t & 63;                   // W2[k][o]
        w2t[o * HS_STRIDE + k] = (__bf16)W2b[t];
    }
    if (tid < HIDDEN)    b1s[tid] = b1[blk * HIDDEN + tid];
    if (tid < BLOCK_DIM) b2s[tid] = b2[blk * BLOCK_DIM + tid];
    __syncthreads();

    // Layer-1 B fragments: this wave's 32 hidden columns, [ntile][kstep].
    // Persist in registers for the whole row loop (32 VGPRs).
    const int cb = wave * 32;
    Frag bW[2][2];
#pragma unroll
    for (int nt = 0; nt < 2; ++nt)
#pragma unroll
        for (int ks = 0; ks < 2; ++ks)
            load_frag(bW[nt][ks], w1t, cb + nt * 16, XS_STRIDE, ks * 32, lr, lh);

    // Layer-2 ownership: wave -> N-tile (wave&3), M-tiles (wave>>2)*2 + {0,1}.
    // All 8 K-step B fragments for this wave's N-tile persist in registers
    // (64 VGPRs), loaded from LDS exactly once per workgroup.
    const int nt2 = wave & 3;
    const int mb2 = (wave >> 2) * 2;
    Frag bW2[8];
#pragma unroll
    for (int ks = 0; ks < 8; ++ks)
        load_frag(bW2[ks], w2t, nt2 * 16, HS_STRIDE, ks * 32, lr, lh);

    const int ocol = nt2 * 16 + lr;           // local output column
    const float bo = b2s[ocol];

    for (int msub = 0; msub < N_SUB; ++msub) {
        const int row0 = mbase + msub * M_TILE;

        // Prefetch next x tile into L2 while we work on this one.
        if (msub + 1 < N_SUB) {
            const float* nxt = x + (size_t)(row0 + M_TILE + (tid & 63)) * IN_DIM
                                 + blk * BLOCK_DIM + ((tid >> 6) << 4);
            __builtin_prefetch(nxt, 0, 0);
        }

        // ---- Load x tile [64 rows][64 cols] f32 -> bf16 LDS ----
        const float* xsrc = x + (size_t)row0 * IN_DIM + blk * BLOCK_DIM;
        for (int j = tid; j < M_TILE * 16; j += 256) {       // 1024 float4
            int r = j >> 4, c4 = (j & 15) << 2;
            float4 vv = *(const float4*)(xsrc + (size_t)r * IN_DIM + c4);
            uint2 pk;
            pk.x = pkbf(vv.x, vv.y);
            pk.y = pkbf(vv.z, vv.w);
            *(uint2*)(xs + r * XS_STRIDE + c4) = pk;         // ds_store_b64
        }
        __syncthreads();

        // ---- Layer 1: [64x64] @ [64x(32 per wave)] -> GELU -> hs ----
        // Double-buffer the two A fragments across M-tiles so the next
        // tile's ds_load_b128s fly under the current tile's GELU chain.
        {
            Frag a0[2], a1[2];
            load_frag(a0[0], xs, 0, XS_STRIDE, 0,  lr, lh);
            load_frag(a1[0], xs, 0, XS_STRIDE, 32, lr, lh);
#pragma unroll
            for (int mt = 0; mt < 4; ++mt) {
                const int buf = mt & 1;
                if (mt + 1 < 4) {
                    load_frag(a0[buf ^ 1], xs, (mt + 1) * 16, XS_STRIDE, 0,  lr, lh);
                    load_frag(a1[buf ^ 1], xs, (mt + 1) * 16, XS_STRIDE, 32, lr, lh);
                }
#pragma unroll
                for (int nt = 0; nt < 2; ++nt) {
                    v8f c = {};
                    c = __builtin_amdgcn_wmma_f32_16x16x32_bf16(
                            false, a0[buf].v, false, bW[nt][0].v, (short)0, c, false, false);
                    c = __builtin_amdgcn_wmma_f32_16x16x32_bf16(
                            false, a1[buf].v, false, bW[nt][1].v, (short)0, c, false, false);
                    const int col = cb + nt * 16 + lr;
                    const float bias = b1s[col];
                    __bf16* hrow = hs + (mt * 16 + lh * 8) * HS_STRIDE + col;
#pragma unroll
                    for (int v = 0; v < 8; ++v)
                        hrow[v * HS_STRIDE] = (__bf16)gelu_exact(c[v] + bias);
                }
            }
        }
        __syncthreads();

        // ---- Layer 2: two 16x16 out tiles per wave, K=256 in 8 WMMA steps ----
#pragma unroll
        for (int mtl = 0; mtl < 2; ++mtl) {
            const int mt = mb2 + mtl;
            v8f c = {};
            Frag a[2];
            load_frag(a[0], hs, mt * 16, HS_STRIDE, 0, lr, lh);
#pragma unroll
            for (int ks = 0; ks < 8; ++ks) {
                if (ks + 1 < 8)
                    load_frag(a[(ks + 1) & 1], hs, mt * 16, HS_STRIDE, (ks + 1) * 32, lr, lh);
                c = __builtin_amdgcn_wmma_f32_16x16x32_bf16(
                        false, a[ks & 1].v, false, bW2[ks].v, (short)0, c, false, false);
            }
            // Single base address; per-element offsets are compile-time
            // constants (v * 16384 B) -> folded into VGLOBAL IOFFSET.
            float* obase = out + (size_t)(row0 + mt * 16 + lh * 8) * IN_DIM
                               + blk * BLOCK_DIM + ocol;
#pragma unroll
            for (int v = 0; v < 8; ++v)
                obase[v * IN_DIM] = c[v] + bo;
        }
        __syncthreads();   // protect xs/hs before next iteration overwrites
    }
}

extern "C" void kernel_launch(void* const* d_in, const int* in_sizes, int n_in,
                              void* d_out, int out_size, void* d_ws, size_t ws_size,
                              hipStream_t stream) {
    const float* x  = (const float*)d_in[0];
    const float* W1 = (const float*)d_in[1];
    const float* b1 = (const float*)d_in[2];
    const float* W2 = (const float*)d_in[3];
    const float* b2 = (const float*)d_in[4];
    float* out = (float*)d_out;

    // >64KB dynamic LDS requires opting in (idempotent, safe every call).
    (void)hipFuncSetAttribute((const void*)blockmlp_wmma_kernel,
                              hipFuncAttributeMaxDynamicSharedMemorySize,
                              SMEM_BYTES);

    dim3 grid(BATCH / M_SUPER, N_BLOCKS);   // 16 x 64 workgroups
    dim3 block(256);                        // 8 waves (wave32)
    blockmlp_wmma_kernel<<<grid, block, SMEM_BYTES, stream>>>(x, W1, b1, W2, b2, out);
}
